// MaskedSimAttn_72447508349116
// MI455X (gfx1250) — matrix operand order repaired and checked by reading, not compiled, not run on hardware
//
#include <hip/hip_runtime.h>
#include <hip/hip_bf16.h>
#include <math.h>

typedef _Float16 f16;
typedef _Float16 v16h __attribute__((ext_vector_type(16)));
typedef _Float16 v8h  __attribute__((ext_vector_type(8)));
typedef float    v8f  __attribute__((ext_vector_type(8)));

#define B_  8
#define L_  1024
#define S_  1024
#define D_  1024
#define H_  16
#define HD_ 64

// ---------------------------------------------------------------------------
// WMMA helpers (CDNA5 wave32, 16x16x32 f16 -> f32)
// ---------------------------------------------------------------------------
__device__ __forceinline__ v8f wmma16(v16h a, v16h b, v8f c) {
  return __builtin_amdgcn_wmma_f32_16x16x32_f16(
      /*neg_a=*/false, a, /*neg_b=*/false, b,
      /*c_mod=*/(short)0, c, /*reuse_a=*/false, /*reuse_b=*/false);
}

// A-fragment (16x32 f16): lane m=lane%16, halves i<8 -> K=g*8+i, i>=8 -> K=16+g*8+i
// Assemble from two contiguous 8-half (16B) chunks.
__device__ __forceinline__ v16h frag2(const f16* p0, const f16* p1) {
  v8h lo = *(const v8h*)p0;
  v8h hi = *(const v8h*)p1;
  v16h r;
#pragma unroll
  for (int i = 0; i < 8; ++i) { r[i] = lo[i]; r[i + 8] = hi[i]; }
  return r;
}

// ---------------------------------------------------------------------------
// Kernel 1: gate = 0.5*(tanh(mask)+1) in f16, permuted into the WMMA C-fragment
// order: gate2[h][l/16][s][l%16]  -> a lane's 8 gate values (fixed s, m=8g..8g+7)
// are one contiguous 16-byte load in the attention hot loop.
// ---------------------------------------------------------------------------
__global__ __launch_bounds__(256) void gate_prep_kernel(
    const float* __restrict__ mask, f16* __restrict__ gate2) {
  size_t i = ((size_t)blockIdx.x * 256 + threadIdx.x) * 4;   // 4 consecutive s
  int h = (int)(i >> 20);
  int rem = (int)(i & ((1u << 20) - 1));
  int l = rem >> 10;
  int s = rem & 1023;
  float4 m = *(const float4*)(mask + i);
  size_t base = (((size_t)h * 64 + (l >> 4)) * S_ + s) * 16 + (l & 15);
  gate2[base + 0 * 16] = (f16)(0.5f * (tanhf(m.x) + 1.0f));
  gate2[base + 1 * 16] = (f16)(0.5f * (tanhf(m.y) + 1.0f));
  gate2[base + 2 * 16] = (f16)(0.5f * (tanhf(m.z) + 1.0f));
  gate2[base + 3 * 16] = (f16)(0.5f * (tanhf(m.w) + 1.0f));
}

// ---------------------------------------------------------------------------
// Projection GEMM: C[M=8192, N=1024] = X f32 @ W f32, result f16.
// Block tile 128x64 (N tile == one head), 8 waves, wave = 16x64 strip.
// MODE 0: Q, store [B,H,L,64]
// MODE 1: K, row-l2norm per head, store [B,H,S,64]
// MODE 2: V, row-l2norm per head, store transposed [B,H,64,S]
// ---------------------------------------------------------------------------
template <int MODE>
__global__ __launch_bounds__(256) void proj_kernel(
    const float* __restrict__ X, const float* __restrict__ W,
    f16* __restrict__ out) {
  __shared__ __align__(64) f16 sA[128][32];   // X tile, row-major [m][k]
  __shared__ __align__(64) f16 sB[64][32];    // W tile transposed [n][k]

  const int mt = blockIdx.x;            // 0..63
  const int nt = blockIdx.y;            // 0..15  (== head index)
  const int tid  = threadIdx.x;
  const int wave = tid >> 5;
  const int lane = tid & 31;
  const int g = lane >> 4;
  const int n = lane & 15;
  const int M0 = mt * 128;
  const int N0 = nt * 64;

  v8f acc[4] = {};

  for (int k0 = 0; k0 < D_; k0 += 32) {
    __syncthreads();
    // X tile: 128x32 f32 -> f16 (1024 float4, 4/thread)
#pragma unroll
    for (int i = 0; i < 4; ++i) {
      int idx = tid + i * 256;
      int r = idx >> 3, c4 = (idx & 7) * 4;
      float4 f = *(const float4*)&X[(size_t)(M0 + r) * D_ + k0 + c4];
      sA[r][c4 + 0] = (f16)f.x; sA[r][c4 + 1] = (f16)f.y;
      sA[r][c4 + 2] = (f16)f.z; sA[r][c4 + 3] = (f16)f.w;
    }
    // W tile: 32x64 f32 -> transposed f16 [n][k] (512 float4, 2/thread)
#pragma unroll
    for (int i = 0; i < 2; ++i) {
      int idx = tid + i * 256;
      int kk = idx >> 4, c4 = (idx & 15) * 4;
      float4 f = *(const float4*)&W[(size_t)(k0 + kk) * D_ + N0 + c4];
      sB[c4 + 0][kk] = (f16)f.x; sB[c4 + 1][kk] = (f16)f.y;
      sB[c4 + 2][kk] = (f16)f.z; sB[c4 + 3][kk] = (f16)f.w;
    }
    __syncthreads();

    const f16* ar = &sA[wave * 16 + n][0];
    v16h a = frag2(ar + g * 8, ar + 16 + g * 8);
#pragma unroll
    for (int t = 0; t < 4; ++t) {
      v16h b = frag2(&sB[t * 16 + n][g * 8], &sB[t * 16 + n][16 + g * 8]);
      acc[t] = wmma16(a, b, acc[t]);
    }
  }

  if constexpr (MODE == 0) {
    // Q: out[((b*H+h)*L + l)*64 + hd]
#pragma unroll
    for (int t = 0; t < 4; ++t)
#pragma unroll
      for (int e = 0; e < 8; ++e) {
        int r = M0 + wave * 16 + e + 8 * g;
        int b = r >> 10, l = r & 1023;
        out[(((size_t)b * H_ + nt) * L_ + l) * HD_ + t * 16 + n] = (f16)acc[t][e];
      }
  } else {
    __shared__ float sCf[128][65];
    __shared__ float sNorm[128];
#pragma unroll
    for (int t = 0; t < 4; ++t)
#pragma unroll
      for (int e = 0; e < 8; ++e)
        sCf[wave * 16 + e + 8 * g][t * 16 + n] = acc[t][e];
    __syncthreads();
    if (tid < 128) {
      float s = 0.f;
#pragma unroll 8
      for (int j = 0; j < 64; ++j) { float x = sCf[tid][j]; s += x * x; }
      sNorm[tid] = fmaxf(sqrtf(s), 1e-12f);
    }
    __syncthreads();
#pragma unroll
    for (int t = 0; t < 4; ++t)
#pragma unroll
      for (int e = 0; e < 8; ++e) {
        int m = wave * 16 + e + 8 * g;
        float val = acc[t][e] / sNorm[m];
        int r = M0 + m;
        int b = r >> 10, sI = r & 1023, hd = t * 16 + n;
        if constexpr (MODE == 1)
          out[(((size_t)b * H_ + nt) * S_ + sI) * HD_ + hd] = (f16)val;
        else
          out[(((size_t)b * H_ + nt) * HD_ + hd) * S_ + sI] = (f16)val;
      }
  }
}

// ---------------------------------------------------------------------------
// Fused gated attention per (b,h): x = (Qh Khᵀ * gate) Vh
// Block = (b, h, 128 L-rows); wave = 16 L-rows; S loop in steps of 32.
// ---------------------------------------------------------------------------
__global__ __launch_bounds__(256) void attn_kernel(
    const f16* __restrict__ qh,     // [B,H,L,64]
    const f16* __restrict__ kh,     // [B,H,S,64]
    const f16* __restrict__ vT,     // [B,H,64,S]
    const f16* __restrict__ gate2,  // [H, L/16, S, 16]  (C-fragment order)
    f16* __restrict__ xout) {       // [B,L,D]
  __shared__ __align__(64) f16 sK[32][64];        // [s][hd]
  __shared__ __align__(64) f16 sV[64][32];        // [hd][s]
  __shared__ __align__(64) f16 sS[8][16][32];     // per-wave gated scores [m][s]

  const int bh = blockIdx.x;            // 0..127
  const int lt = blockIdx.y;            // 0..7
  const int b = bh >> 4, h = bh & 15;
  const int tid  = threadIdx.x;
  const int wave = tid >> 5;
  const int lane = tid & 31;
  const int g = lane >> 4;
  const int n = lane & 15;

  const size_t baseQ = ((size_t)bh * L_ + lt * 128) * HD_;
  const size_t baseK = (size_t)bh * S_ * HD_;
  const size_t baseV = (size_t)bh * HD_ * S_;
  // gate2 base for this wave's 16-row block; lane offset (s*16 + 8g) added per step
  const f16* gwave = gate2 + (((size_t)h * 64 + lt * 8 + wave) * S_) * 16 + 8 * g;

  // Q A-fragments for this wave's 16 rows (kept in registers for whole S loop)
  const f16* qrow = qh + baseQ + (size_t)(wave * 16 + n) * HD_;
  v16h qa[2];
  qa[0] = frag2(qrow + g * 8,      qrow + 16 + g * 8);
  qa[1] = frag2(qrow + 32 + g * 8, qrow + 48 + g * 8);

  v8f xacc[4] = {};

  for (int s0 = 0; s0 < S_; s0 += 32) {
    __syncthreads();
    { // K tile: 32 rows x 64 halves (16B chunk per thread)
      int r = tid >> 3, c = (tid & 7) * 8;
      *(v8h*)&sK[r][c] = *(const v8h*)&kh[baseK + (size_t)(s0 + r) * HD_ + c];
    }
    { // Vᵀ tile: 64 rows x 32 halves (16B chunk per thread)
      int r = tid >> 2, c = (tid & 3) * 8;
      *(v8h*)&sV[r][c] = *(const v8h*)&vT[baseV + (size_t)r * S_ + s0 + c];
    }
    __syncthreads();

    // scores = Q Khᵀ over HD=64 (2 k-steps), then gate, stash in LDS as f16
#pragma unroll
    for (int st = 0; st < 2; ++st) {
      v8f sc = {};
#pragma unroll
      for (int kk = 0; kk < 2; ++kk) {
        v16h bfr = frag2(&sK[st * 16 + n][kk * 32 + g * 8], &sK[st * 16 + n][kk * 32 + 16 + g * 8]);
        sc = wmma16(qa[kk], bfr, sc);
      }
      // one contiguous 16B gate load covers this lane's 8 elements
      v8h gv = *(const v8h*)&gwave[(size_t)(s0 + st * 16 + n) * 16];
#pragma unroll
      for (int e = 0; e < 8; ++e)
        sS[wave][e + 8 * g][st * 16 + n] = (f16)(sc[e] * (float)gv[e]);
    }
    // wave-private LDS re-layout: in-wave DS ordering suffices (no barrier)
    const f16* srow = &sS[wave][n][0];
    v16h sa = frag2(srow + g * 8, srow + 16 + g * 8);
#pragma unroll
    for (int jt = 0; jt < 4; ++jt) {
      v16h bfr = frag2(&sV[jt * 16 + n][g * 8], &sV[jt * 16 + n][16 + g * 8]);
      xacc[jt] = wmma16(sa, bfr, xacc[jt]);
    }
  }

  // write f16 heads output into [B, L, D] (col = h*64 + hd)
#pragma unroll
  for (int jt = 0; jt < 4; ++jt)
#pragma unroll
    for (int e = 0; e < 8; ++e) {
      int l = lt * 128 + wave * 16 + e + 8 * g;
      int col = h * 64 + jt * 16 + n;
      xout[((size_t)b * L_ + l) * D_ + col] = (f16)xacc[jt][e];
    }
}

// ---------------------------------------------------------------------------
// Output projection: out f32 = Xattn f16 [8192,1024] @ Wo f32
// ---------------------------------------------------------------------------
__global__ __launch_bounds__(256) void gemm_out_kernel(
    const f16* __restrict__ X, const float* __restrict__ W,
    float* __restrict__ out) {
  __shared__ __align__(64) f16 sA[128][32];
  __shared__ __align__(64) f16 sB[64][32];

  const int mt = blockIdx.x, nt = blockIdx.y;
  const int tid  = threadIdx.x;
  const int wave = tid >> 5;
  const int lane = tid & 31;
  const int g = lane >> 4;
  const int n = lane & 15;
  const int M0 = mt * 128, N0 = nt * 64;

  v8f acc[4] = {};

  for (int k0 = 0; k0 < D_; k0 += 32) {
    __syncthreads();
#pragma unroll
    for (int i = 0; i < 2; ++i) {   // A tile already f16: 512 16B chunks
      int idx = tid + i * 256;
      int r = idx >> 2, c = (idx & 3) * 8;
      *(v8h*)&sA[r][c] = *(const v8h*)&X[(size_t)(M0 + r) * D_ + k0 + c];
    }
#pragma unroll
    for (int i = 0; i < 2; ++i) {   // W tile -> transposed f16
      int idx = tid + i * 256;
      int kk = idx >> 4, c4 = (idx & 15) * 4;
      float4 f = *(const float4*)&W[(size_t)(k0 + kk) * D_ + N0 + c4];
      sB[c4 + 0][kk] = (f16)f.x; sB[c4 + 1][kk] = (f16)f.y;
      sB[c4 + 2][kk] = (f16)f.z; sB[c4 + 3][kk] = (f16)f.w;
    }
    __syncthreads();

    const f16* ar = &sA[wave * 16 + n][0];
    v16h a = frag2(ar + g * 8, ar + 16 + g * 8);
#pragma unroll
    for (int t = 0; t < 4; ++t) {
      v16h b = frag2(&sB[t * 16 + n][g * 8], &sB[t * 16 + n][16 + g * 8]);
      acc[t] = wmma16(a, b, acc[t]);
    }
  }

#pragma unroll
  for (int t = 0; t < 4; ++t)
#pragma unroll
    for (int e = 0; e < 8; ++e) {
      int r = M0 + wave * 16 + e + 8 * g;
      out[(size_t)r * D_ + N0 + t * 16 + n] = acc[t][e];
    }
}

// ---------------------------------------------------------------------------
extern "C" void kernel_launch(void* const* d_in, const int* in_sizes, int n_in,
                              void* d_out, int out_size, void* d_ws, size_t ws_size,
                              hipStream_t stream) {
  (void)in_sizes; (void)n_in; (void)out_size; (void)ws_size;
  const float* q    = (const float*)d_in[0];
  const float* k    = (const float*)d_in[1];
  const float* v    = (const float*)d_in[2];
  const float* Wq   = (const float*)d_in[3];
  const float* Wk   = (const float*)d_in[4];
  const float* Wv   = (const float*)d_in[5];
  const float* Wo   = (const float*)d_in[6];
  const float* mask = (const float*)d_in[7];
  float* out = (float*)d_out;

  char* ws = (char*)d_ws;
  const size_t nHead = (size_t)B_ * H_ * L_ * HD_;   // 8.39M elems
  const size_t nGate = (size_t)H_ * L_ * S_;         // 16.8M elems
  f16* qh    = (f16*)ws;                      ws += nHead * sizeof(f16);
  f16* kh    = (f16*)ws;                      ws += nHead * sizeof(f16);
  f16* vT    = (f16*)ws;                      ws += nHead * sizeof(f16);
  f16* gate2 = (f16*)ws;                      ws += nGate * sizeof(f16);
  f16* xattn = (f16*)ws;

  // 1) gate = 0.5*(tanh(mask)+1), permuted to C-fragment order [H,L/16,S,16]
  gate_prep_kernel<<<dim3((unsigned)(nGate / 1024)), 256, 0, stream>>>(mask, gate2);
  // 2) projections (M=8192 rows, 64x16 tiles of 128x64)
  proj_kernel<0><<<dim3(64, 16), 256, 0, stream>>>(q, Wq, qh);
  proj_kernel<1><<<dim3(64, 16), 256, 0, stream>>>(k, Wk, kh);
  proj_kernel<2><<<dim3(64, 16), 256, 0, stream>>>(v, Wv, vT);
  // 3) fused gated attention
  attn_kernel<<<dim3(B_ * H_, L_ / 128), 256, 0, stream>>>(qh, kh, vT, gate2, xattn);
  // 4) output projection
  gemm_out_kernel<<<dim3(64, 16), 256, 0, stream>>>(xattn, Wo, out);
}
